// MultiHeadAttention_14903536517189
// MI455X (gfx1250) — matrix operand order repaired and checked by reading, not compiled, not run on hardware
//
#include <hip/hip_runtime.h>

#define D_MODEL 1024
#define N_HEADS 16
#define D_K     64
#define T_SEQ   2048
#define B_BATCH 2

typedef __attribute__((ext_vector_type(16))) __bf16 v16bf;
typedef __attribute__((ext_vector_type(8)))  float  v8f;

union FragAB { v16bf v; uint4 q[2]; };

__device__ __forceinline__ unsigned short f32_to_bf16(float f) {
    unsigned int u = __float_as_uint(f);
    unsigned int r = u + 0x7FFFu + ((u >> 16) & 1u);   // round-nearest-even
    return (unsigned short)(r >> 16);
}

__device__ __forceinline__ unsigned pack_bf16x2(float lo, float hi) {
    return (unsigned)f32_to_bf16(lo) | ((unsigned)f32_to_bf16(hi) << 16);
}

// gfx1250 async copy: 16B global -> LDS, tracked by ASYNCcnt.
__device__ __forceinline__ void async_g2l_b128(unsigned ldsOff, const void* gaddr) {
    asm volatile("global_load_async_to_lds_b128 %0, %1, off"
                 :: "v"(ldsOff), "v"(gaddr) : "memory");
}
__device__ __forceinline__ void wait_asynccnt0() {
    asm volatile("s_wait_asynccnt 0" ::: "memory");
}
__device__ __forceinline__ unsigned lds_off32(const void* p) {
    return (unsigned)(size_t)p;     // low 32 bits of flat shared ptr = LDS offset
}

// ---------------------------------------------------------------------------
// C[64x64 tile] = A[M x 1024] * W[1024 x 1024] + bias, double-buffered LDS,
// global loads pipelined two k-tiles ahead through registers.
// OUT_MODE: 0 -> fp32 [M][1024]                  (final output)
//           1 -> bf16 [B][H][T][dk] head layout  (Q, K)
//           2 -> bf16 [B][H][dk][T] transposed   (V, pre-transposed for PV WMMA)
// ---------------------------------------------------------------------------
template <bool A_BF16, int OUT_MODE>
__global__ __launch_bounds__(128)
void gemm_bias_64x64(const void* __restrict__ Aptr,
                     const float* __restrict__ W,
                     const float* __restrict__ bias,
                     void* __restrict__ Out) {
    __shared__ alignas(16) unsigned short a_lds[2][64 * 32];   // [row][k]
    __shared__ alignas(16) unsigned short b_lds[2][64 * 32];   // [col][k] (transposed)

    const int t    = threadIdx.x;
    const int lane = t & 31;
    const int wv   = t >> 5;        // wave 0..3 -> rows wv*16..+15
    const int hi   = lane >> 4;     // lane half (K-group for A, K-half for B)
    const int l15  = lane & 15;
    const int rowBase = blockIdx.y * 64;
    const int nBase   = blockIdx.x * 64;

    v8f acc[4];
    #pragma unroll
    for (int nt = 0; nt < 4; ++nt)
        #pragma unroll
        for (int j = 0; j < 8; ++j) acc[nt][j] = 0.0f;

    const float*          Af = (const float*)Aptr;
    const unsigned short* Ab = (const unsigned short*)Aptr;

    // per-thread staging registers for one k-tile
    unsigned aP[8];
    float    bf0[8], bf1[8];

    auto loadTile = [&](int kb) {
        #pragma unroll
        for (int i = 0; i < 8; ++i) {
            int idx = i * 128 + t;              // pair index, 1024 total
            int r = idx >> 4, kp = idx & 15;
            if (A_BF16) {
                aP[i] = *(const unsigned*)(Ab + (size_t)(rowBase + r) * 1024 + kb + kp * 2);
            } else {
                const float* s = Af + (size_t)(rowBase + r) * 1024 + kb + kp * 2;
                aP[i] = pack_bf16x2(s[0], s[1]);
            }
        }
        #pragma unroll
        for (int i = 0; i < 8; ++i) {
            int idx = i * 128 + t;
            int n = idx & 63, kp = idx >> 6;
            bf0[i] = W[(size_t)(kb + kp * 2)     * 1024 + nBase + n];
            bf1[i] = W[(size_t)(kb + kp * 2 + 1) * 1024 + nBase + n];
        }
    };
    auto storeTile = [&](int sel) {
        #pragma unroll
        for (int i = 0; i < 8; ++i) {
            int idx = i * 128 + t;
            int r = idx >> 4, kp = idx & 15;
            ((unsigned*)a_lds[sel])[r * 16 + kp] = aP[i];
        }
        #pragma unroll
        for (int i = 0; i < 8; ++i) {
            int idx = i * 128 + t;
            int n = idx & 63, kp = idx >> 6;
            ((unsigned*)b_lds[sel])[n * 16 + kp] = pack_bf16x2(bf0[i], bf1[i]);
        }
    };

    loadTile(0);
    storeTile(0);
    loadTile(32);
    __syncthreads();                            // tile 0 visible

    for (int it = 0; it < 32; ++it) {
        const int cur = it & 1;
        if (it + 1 < 32) storeTile(cur ^ 1);    // tile it+1 -> other buffer
        if (it + 2 < 32) loadTile((it + 2) * 32);

        const unsigned short* ab = a_lds[cur];
        const unsigned short* bb = b_lds[cur];
        // A fragment: lane holds row, K elems {hi*8..+7, 16+hi*8..+7}
        FragAB afr;
        {
            int row = wv * 16 + l15;
            afr.q[0] = *(const uint4*)&ab[row * 32 + hi * 8];
            afr.q[1] = *(const uint4*)&ab[row * 32 + 16 + hi * 8];
        }
        #pragma unroll
        for (int nt = 0; nt < 4; ++nt) {
            // B fragment: lane holds column, K elems {hi*8..+7, 16+hi*8..+7}
            FragAB bfr;
            int col = nt * 16 + l15;
            bfr.q[0] = *(const uint4*)&bb[col * 32 + hi * 8];
            bfr.q[1] = *(const uint4*)&bb[col * 32 + 16 + hi * 8];
            acc[nt] = __builtin_amdgcn_wmma_f32_16x16x32_bf16(
                false, afr.v, false, bfr.v, (short)0, acc[nt], false, false);
        }
        __syncthreads();   // all reads of buf cur done; tile it+1 stores visible
    }

    // epilogue: C layout is (VGPR r, lane) = (row r + 8*hi, col l15)
    #pragma unroll
    for (int nt = 0; nt < 4; ++nt) {
        int col = nBase + nt * 16 + l15;
        float bv = bias[col];
        #pragma unroll
        for (int r = 0; r < 8; ++r) {
            int row = rowBase + wv * 16 + r + hi * 8;
            float val = acc[nt][r] + bv;
            int b = row >> 11, tt = row & (T_SEQ - 1);
            int h = col >> 6,  d  = col & 63;
            if (OUT_MODE == 0) {
                ((float*)Out)[(size_t)row * 1024 + col] = val;
            } else if (OUT_MODE == 1) {
                ((unsigned short*)Out)[(((size_t)(b * N_HEADS + h) * T_SEQ + tt) * D_K) + d]
                    = f32_to_bf16(val);
            } else {
                ((unsigned short*)Out)[(((size_t)(b * N_HEADS + h) * D_K + d) * T_SEQ) + tt]
                    = f32_to_bf16(val);
            }
        }
    }
}

// ---------------------------------------------------------------------------
// Flash attention: one WG per (b, h, 64-query block); 4 waves x 16 queries.
// Double-buffered K/V tiles fetched with global_load_async_to_lds_b128 so the
// next tile's DMA overlaps the current tile's 16 WMMAs + softmax.
// ---------------------------------------------------------------------------
__global__ __launch_bounds__(128)
void flash_attn(const unsigned short* __restrict__ Qh,
                const unsigned short* __restrict__ Kh,
                const unsigned short* __restrict__ Vt,
                unsigned short* __restrict__ Ctx) {
    __shared__ alignas(16) unsigned short k_lds[2][64 * 64];   // [key][dk]
    __shared__ alignas(16) unsigned short v_lds[2][64 * 64];   // [dk][key]
    __shared__ alignas(16) unsigned short p_lds[4][16 * 64];   // per-wave P scratch

    const int t    = threadIdx.x;
    const int lane = t & 31;
    const int wv   = t >> 5;
    const int hi   = lane >> 4;
    const int l15  = lane & 15;
    const int b = blockIdx.z, h = blockIdx.y;
    const int qBase = blockIdx.x * 64;
    const size_t headOff = (size_t)(b * N_HEADS + h) * T_SEQ * D_K;

    // Q fragments held in registers for the whole key loop (2 x 16x32 bf16)
    FragAB qf[2];
    {
        int q = qBase + wv * 16 + l15;
        const unsigned short* qrow = Qh + headOff + (size_t)q * D_K;
        #pragma unroll
        for (int s = 0; s < 2; ++s) {
            qf[s].q[0] = *(const uint4*)(qrow + s * 32 + hi * 8);
            qf[s].q[1] = *(const uint4*)(qrow + s * 32 + 16 + hi * 8);
        }
    }

    v8f oacc[4];
    #pragma unroll
    for (int nt = 0; nt < 4; ++nt)
        #pragma unroll
        for (int j = 0; j < 8; ++j) oacc[nt][j] = 0.0f;
    float rowMax[8], rowSum[8];
    #pragma unroll
    for (int r = 0; r < 8; ++r) { rowMax[r] = -3.0e38f; rowSum[r] = 0.0f; }

    const unsigned kldsBase = lds_off32(k_lds);
    const unsigned vldsBase = lds_off32(v_lds);

    auto issueTile = [&](int kt, int sel) {
        // K tile: contiguous 8KB (4 x b128 per thread)
        const unsigned short* ksrc = Kh + headOff + (size_t)kt * D_K;
        unsigned kb = kldsBase + sel * (64 * 64 * 2);
        #pragma unroll
        for (int i = 0; i < 4; ++i) {
            int idx = i * 128 + t;              // 16B chunk index, 512 total
            async_g2l_b128(kb + idx * 16, ksrc + idx * 8);
        }
        // V tile: 64 dk-rows of 64 keys (row stride T_SEQ)
        const unsigned short* vsrc = Vt + (size_t)(b * N_HEADS + h) * D_K * T_SEQ + kt;
        unsigned vb = vldsBase + sel * (64 * 64 * 2);
        #pragma unroll
        for (int i = 0; i < 4; ++i) {
            int idx = i * 128 + t;
            int d = idx >> 3, c8 = idx & 7;
            async_g2l_b128(vb + (d * 64 + c8 * 8) * 2, vsrc + (size_t)d * T_SEQ + c8 * 8);
        }
    };

    issueTile(0, 0);

    for (int it = 0; it < T_SEQ / 64; ++it) {
        const int cur = it & 1;
        wait_asynccnt0();                       // my copies for tile it landed
        __syncthreads();                        // all copies visible; other buf free
        if (it + 1 < T_SEQ / 64) issueTile((it + 1) * 64, cur ^ 1);

        const unsigned short* kl = k_lds[cur];
        const unsigned short* vl = v_lds[cur];

        // S = Q * K^T  (4 key subtiles x 2 K-steps over dk)
        v8f sacc[4];
        #pragma unroll
        for (int st = 0; st < 4; ++st) {
            #pragma unroll
            for (int j = 0; j < 8; ++j) sacc[st][j] = 0.0f;
            #pragma unroll
            for (int s = 0; s < 2; ++s) {
                FragAB kf;                      // B frag: lane = key column
                int key = st * 16 + l15;
                kf.q[0] = *(const uint4*)&kl[key * 64 + s * 32 + hi * 8];
                kf.q[1] = *(const uint4*)&kl[key * 64 + s * 32 + 16 + hi * 8];
                sacc[st] = __builtin_amdgcn_wmma_f32_16x16x32_bf16(
                    false, qf[s].v, false, kf.v, (short)0, sacc[st], false, false);
            }
        }

        // online softmax; rows r+8*hi live across the 16-lane half -> xor<=8 shuffles
        unsigned short pb[4][8];
        #pragma unroll
        for (int r = 0; r < 8; ++r) {
            float mx = -3.0e38f;
            #pragma unroll
            for (int st = 0; st < 4; ++st) {
                float sv = sacc[st][r] * 0.125f;   // 1/sqrt(dk)
                sacc[st][r] = sv;
                mx = fmaxf(mx, sv);
            }
            mx = fmaxf(mx, __shfl_xor(mx, 1, 32));
            mx = fmaxf(mx, __shfl_xor(mx, 2, 32));
            mx = fmaxf(mx, __shfl_xor(mx, 4, 32));
            mx = fmaxf(mx, __shfl_xor(mx, 8, 32));
            float mNew = fmaxf(rowMax[r], mx);
            float corr = __expf(rowMax[r] - mNew);
            float psum = 0.0f;
            #pragma unroll
            for (int st = 0; st < 4; ++st) {
                float p = __expf(sacc[st][r] - mNew);
                psum += p;
                pb[st][r] = f32_to_bf16(p);
            }
            psum += __shfl_xor(psum, 1, 32);
            psum += __shfl_xor(psum, 2, 32);
            psum += __shfl_xor(psum, 4, 32);
            psum += __shfl_xor(psum, 8, 32);
            rowSum[r] = rowSum[r] * corr + psum;
            rowMax[r] = mNew;
            #pragma unroll
            for (int nt = 0; nt < 4; ++nt) oacc[nt][r] *= corr;
        }

        // C-layout -> A-layout transpose of P via per-wave LDS scratch
        unsigned short* pl = &p_lds[wv][0];
        #pragma unroll
        for (int r = 0; r < 8; ++r) {
            int m = r + hi * 8;
            #pragma unroll
            for (int st = 0; st < 4; ++st)
                pl[m * 64 + st * 16 + l15] = pb[st][r];
        }
        // cross-lane LDS dependency invisible to compiler: drain DS before reload
        asm volatile("s_wait_dscnt 0" ::: "memory");

        // O += P * V  (2 K-steps over keys x 4 dk subtiles)
        #pragma unroll
        for (int ks = 0; ks < 2; ++ks) {
            FragAB pf;
            pf.q[0] = *(const uint4*)&pl[l15 * 64 + ks * 32 + hi * 8];
            pf.q[1] = *(const uint4*)&pl[l15 * 64 + ks * 32 + 16 + hi * 8];
            #pragma unroll
            for (int nt = 0; nt < 4; ++nt) {
                FragAB vf;                      // B frag: lane = dk column
                int d = nt * 16 + l15;
                vf.q[0] = *(const uint4*)&vl[d * 64 + ks * 32 + hi * 8];
                vf.q[1] = *(const uint4*)&vl[d * 64 + ks * 32 + 16 + hi * 8];
                oacc[nt] = __builtin_amdgcn_wmma_f32_16x16x32_bf16(
                    false, pf.v, false, vf.v, (short)0, oacc[nt], false, false);
            }
        }
    }

    // normalize, write ctx bf16 in [B][T][D_MODEL] for the O-projection GEMM
    #pragma unroll
    for (int r = 0; r < 8; ++r) {
        float inv = 1.0f / rowSum[r];
        int qrow = qBase + wv * 16 + r + hi * 8;
        #pragma unroll
        for (int nt = 0; nt < 4; ++nt) {
            float val = oacc[nt][r] * inv;
            Ctx[((size_t)(b * T_SEQ) + qrow) * D_MODEL + h * 64 + nt * 16 + l15]
                = f32_to_bf16(val);
        }
    }
}

// ---------------------------------------------------------------------------
extern "C" void kernel_launch(void* const* d_in, const int* in_sizes, int n_in,
                              void* d_out, int out_size, void* d_ws, size_t ws_size,
                              hipStream_t stream) {
    const float* q  = (const float*)d_in[0];
    const float* k  = (const float*)d_in[1];
    const float* v  = (const float*)d_in[2];
    const float* Wq = (const float*)d_in[3];
    const float* bq = (const float*)d_in[4];
    const float* Wk = (const float*)d_in[5];
    const float* bk = (const float*)d_in[6];
    const float* Wv = (const float*)d_in[7];
    const float* bv = (const float*)d_in[8];
    const float* Wo = (const float*)d_in[9];
    const float* bo = (const float*)d_in[10];

    const size_t nElem = (size_t)B_BATCH * T_SEQ * D_MODEL;   // 4M bf16 each
    unsigned short* qh  = (unsigned short*)d_ws;
    unsigned short* kh  = qh + nElem;
    unsigned short* vt  = kh + nElem;
    unsigned short* ctx = vt + nElem;

    dim3 blk(128);
    dim3 gGemm(D_MODEL / 64, (B_BATCH * T_SEQ) / 64);
    gemm_bias_64x64<false, 1><<<gGemm, blk, 0, stream>>>(q, Wq, bq, qh);
    gemm_bias_64x64<false, 1><<<gGemm, blk, 0, stream>>>(k, Wk, bk, kh);
    gemm_bias_64x64<false, 2><<<gGemm, blk, 0, stream>>>(v, Wv, bv, vt);

    dim3 gAttn(T_SEQ / 64, N_HEADS, B_BATCH);
    flash_attn<<<gAttn, blk, 0, stream>>>(qh, kh, vt, ctx);

    gemm_bias_64x64<true, 0><<<gGemm, blk, 0, stream>>>(ctx, Wo, bo, d_out);
}
